// SAFF_38740605010389
// MI455X (gfx1250) — compile-verified, matrix operand order checked by reading, not run on hardware
//
#include <hip/hip_runtime.h>
#include <hip/hip_bf16.h>

// ---------------------------------------------------------------------------
// Deformable-DETR encoder layer for MI455X (gfx1250, wave32, WMMA).
// GEMMs: v_wmma_f32_16x16x32_bf16, double-buffered LDS staging via
// GLOBAL_LOAD_ASYNC_TO_LDS_B128 (ASYNCcnt-tracked) on CDNA5.
// ---------------------------------------------------------------------------

typedef __attribute__((ext_vector_type(16))) __bf16 v16bf;
typedef __attribute__((ext_vector_type(8)))  __bf16 v8bf;
typedef __attribute__((ext_vector_type(8)))  float  v8f;
typedef __attribute__((ext_vector_type(4)))  int    v4i;

#define BM 64
#define BN 64
#define BK 32

#if defined(__AMDGCN__) && __has_builtin(__builtin_amdgcn_global_load_async_to_lds_b128)
#define USE_ASYNC_LDS 1
#endif

#if defined(USE_ASYNC_LDS)
#if __has_builtin(__builtin_amdgcn_s_wait_asynccnt)
#define WAIT_ASYNC(n) __builtin_amdgcn_s_wait_asynccnt(n)
#else
#define WAIT_ASYNC(n) asm volatile("s_wait_asynccnt %0" ::"n"(n) : "memory")
#endif
#else
#define WAIT_ASYNC(n) ((void)0)
#endif

// 16-byte global -> LDS copy (async on CDNA5, sync fallback elsewhere)
__device__ __forceinline__ void cp16_g2l(const __bf16* g, __bf16* l) {
#if defined(USE_ASYNC_LDS)
  __builtin_amdgcn_global_load_async_to_lds_b128(
      (__attribute__((address_space(1))) v4i*)(const void*)g,
      (__attribute__((address_space(3))) v4i*)(void*)l, 0, 0);
#else
  *(v8bf*)l = *(const v8bf*)g;
#endif
}

// ---------------------------------------------------------------------------
// Generic WMMA GEMM: C[M x Nc] = A[M x K] * B[Nc x K]^T + bias.
// Optional activation, bf16 copy of C, NCHW transposed store, and per-level
// token row mapping (arow = (m/Pl)*Ltot + tokBase + m%Pl; identity case is
// handled by the host passing Pl=M, Ltot=0, tokBase=0 -> no branch).
// Out-of-range rows/cols are index-CLAMPED (their products land in C tiles
// that are never stored), so the staging loop has no exec-mask branches.
// Block: 128 threads = 4 waves (2x2); wave owns a 32x32 C tile (2x2 WMMAs).
// Double-buffered 64x32 LDS tiles.
// ---------------------------------------------------------------------------
__launch_bounds__(128)
__global__ void gemm_wmma_bf16(const __bf16* __restrict__ A,
                               const __bf16* __restrict__ B,
                               const float*  __restrict__ bias,
                               float*        __restrict__ C32,
                               __bf16*       __restrict__ Cbf,
                               float*        __restrict__ Cn,
                               int M, int Nc, int K, int act,
                               int Pl, long tokBase, long Ltot) {
  __shared__ __align__(16) __bf16 As[2][BM * BK];
  __shared__ __align__(16) __bf16 Bs[2][BN * BK];

  const int tid  = threadIdx.x;
  const int wave = tid >> 5;
  const int lane = tid & 31;
  const int wm = wave >> 1;  // 0..1
  const int wn = wave & 1;   // 0..1
  const int mBlock = blockIdx.x * BM;
  const int nBlock = blockIdx.y * BN;

  // ---- per-thread staging assignment: 2 A chunks + 2 B chunks (16B each) --
  const int c0 = tid, c1 = tid + 128;            // chunks 0..255
  const int aRow0 = c0 >> 2, aKk0 = (c0 & 3) << 3;
  const int aRow1 = c1 >> 2, aKk1 = (c1 & 3) << 3;
  const int gm0 = min(mBlock + aRow0, M - 1);
  const int gm1 = min(mBlock + aRow1, M - 1);
  const long ar0 = ((long)(gm0 / Pl)) * Ltot + tokBase + (gm0 % Pl);
  const long ar1 = ((long)(gm1 / Pl)) * Ltot + tokBase + (gm1 % Pl);
  const __bf16* aG0 = A + ar0 * K + aKk0;
  const __bf16* aG1 = A + ar1 * K + aKk1;
  const int aOff0 = aRow0 * BK + aKk0;
  const int aOff1 = aRow1 * BK + aKk1;

  const int gn0 = min(nBlock + aRow0, Nc - 1);
  const int gn1 = min(nBlock + aRow1, Nc - 1);
  const __bf16* bG0 = B + (long)gn0 * K + aKk0;
  const __bf16* bG1 = B + (long)gn1 * K + aKk1;
  const int bOff0 = aOff0;
  const int bOff1 = aOff1;

  v8f acc[2][2];
#pragma unroll
  for (int i = 0; i < 2; ++i)
#pragma unroll
    for (int j = 0; j < 2; ++j)
#pragma unroll
      for (int e = 0; e < 8; ++e) acc[i][j][e] = 0.f;

  const int nk = K / BK;

  // prologue: stage tile 0 into buffer 0
  {
    cp16_g2l(aG0, &As[0][aOff0]);
    cp16_g2l(aG1, &As[0][aOff1]);
    cp16_g2l(bG0, &Bs[0][bOff0]);
    cp16_g2l(bG1, &Bs[0][bOff1]);
  }

  for (int kt = 0; kt < nk; ++kt) {
    const int cur = kt & 1;
    const bool pre = (kt + 1) < nk;
    if (pre) {  // prefetch next tile into the other buffer
      const long ko = (long)(kt + 1) * BK;
      cp16_g2l(aG0 + ko, &As[cur ^ 1][aOff0]);
      cp16_g2l(aG1 + ko, &As[cur ^ 1][aOff1]);
      cp16_g2l(bG0 + ko, &Bs[cur ^ 1][bOff0]);
      cp16_g2l(bG1 + ko, &Bs[cur ^ 1][bOff1]);
      WAIT_ASYNC(4);  // in-order completion => tile kt fully in LDS
    } else {
      WAIT_ASYNC(0);
    }
    __syncthreads();

    // ---- fragments per ISA 16-bit layout:
    // lane<16 : K = {0..7, 16..23};  lane>=16 : K = {8..15, 24..31}
    const int khalf = (lane < 16) ? 0 : 8;
    const int rA = wm * 32 + (lane & 15);
    const int rB = wn * 32 + (lane & 15);
    v16bf afrag[2], bfrag[2];
#pragma unroll
    for (int t = 0; t < 2; ++t) {
      const __bf16* pa = &As[cur][(rA + t * 16) * BK + khalf];
      v8bf alo = *(const v8bf*)pa;
      v8bf ahi = *(const v8bf*)(pa + 16);
      const __bf16* pb = &Bs[cur][(rB + t * 16) * BK + khalf];
      v8bf blo = *(const v8bf*)pb;
      v8bf bhi = *(const v8bf*)(pb + 16);
#pragma unroll
      for (int e = 0; e < 8; ++e) {
        afrag[t][e] = alo[e]; afrag[t][8 + e] = ahi[e];
        bfrag[t][e] = blo[e]; bfrag[t][8 + e] = bhi[e];
      }
    }
#pragma unroll
    for (int i = 0; i < 2; ++i)
#pragma unroll
      for (int j = 0; j < 2; ++j)
        acc[i][j] = __builtin_amdgcn_wmma_f32_16x16x32_bf16(
            false, afrag[i], false, bfrag[j], (short)0, acc[i][j], false, false);
    __syncthreads();  // everyone done reading buf[cur] before it is re-staged
  }

  // ---- epilogue: D element e lives at M = e + 8*(lane>=16), N = lane&15 ----
  const int nLane = lane & 15;
  const int mOff  = (lane >> 4) << 3;
#pragma unroll
  for (int i = 0; i < 2; ++i)
#pragma unroll
    for (int j = 0; j < 2; ++j) {
      const int nCol = nBlock + wn * 32 + j * 16 + nLane;
      if (nCol >= Nc) continue;
      const float bv = bias ? bias[nCol] : 0.f;
#pragma unroll
      for (int e = 0; e < 8; ++e) {
        const int mRow = mBlock + wm * 32 + i * 16 + mOff + e;
        if (mRow >= M) continue;
        float v = acc[i][j][e] + bv;
        if (act == 1)      v = fmaxf(v, 0.f);
        else if (act == 2) v = 1.f / (1.f + __expf(-v));
        if (C32) C32[(long)mRow * Nc + nCol] = v;
        if (Cbf) Cbf[(long)mRow * Nc + nCol] = (__bf16)v;
        if (Cn) {
          const int nb = mRow / Pl, pp = mRow % Pl;
          Cn[((long)nb * Nc + nCol) * Pl + pp] = v;
        }
      }
    }
}

// ---------------------------------------------------------------------------
// fp32 -> bf16 convert (weights)
// ---------------------------------------------------------------------------
__global__ void cvt_f32_bf16(const float* __restrict__ x, __bf16* __restrict__ y, long n) {
  long i = (long)blockIdx.x * 256 + threadIdx.x;
  if (i < n) y[i] = (__bf16)x[i];
}

// NCHW fp32 -> [N*P x C] bf16 (token-major conv GEMM input)
__global__ void nchw_to_tok_bf16(const float* __restrict__ x, __bf16* __restrict__ out,
                                 int C, int P, long total) {
  long i = (long)blockIdx.x * 256 + threadIdx.x;
  if (i >= total) return;
  int c = (int)(i % C);
  int p = (int)((i / C) % P);
  int n = (int)(i / ((long)C * P));
  out[i] = (__bf16)x[((long)n * C + c) * P + p];
}

// level-1 (x1) straight into src (fp32 + bf16), token-major
__global__ void copy_x1_to_src(const float* __restrict__ x1, float* __restrict__ s32,
                               __bf16* __restrict__ sbf, long tokBase, long Ltot, long total) {
  long i = (long)blockIdx.x * 256 + threadIdx.x;
  if (i >= total) return;
  int c = (int)(i % 256);
  int p = (int)((i / 256) % 2500);
  int n = (int)(i / (256L * 2500));
  float v = x1[((long)n * 256 + c) * 2500 + p];
  long dst = ((long)n * Ltot + tokBase + p) * 256 + c;
  s32[dst] = v;
  sbf[dst] = (__bf16)v;
}

// ---------------------------------------------------------------------------
// Direct 3x3 stride-2 pad-1 conv on x3 (only 98 output pixels) -> token-major
// ---------------------------------------------------------------------------
__global__ void conv3x3_s2(const float* __restrict__ x, const float* __restrict__ W,
                           const float* __restrict__ bias, float* __restrict__ y) {
  const int pix = blockIdx.x;            // 0..97
  const int n = pix / 49, p = pix % 49;
  const int oh = p / 7, ow = p % 7;
  const int co = threadIdx.x;            // 0..255
  float acc = bias[co];
  for (int ci = 0; ci < 1024; ++ci) {
#pragma unroll
    for (int kh = 0; kh < 3; ++kh) {
      const int ih = oh * 2 - 1 + kh;
      if (ih < 0 || ih >= 13) continue;
#pragma unroll
      for (int kw = 0; kw < 3; ++kw) {
        const int iw = ow * 2 - 1 + kw;
        if (iw < 0 || iw >= 13) continue;
        acc += W[(((long)co * 1024 + ci) * 3 + kh) * 3 + kw] *
               x[(((long)n * 1024 + ci) * 13 + ih) * 13 + iw];
      }
    }
  }
  y[((long)n * 49 + p) * 256 + co] = acc;
}

// ---------------------------------------------------------------------------
// GroupNorm over a level's token-major conv output -> src (fp32 + bf16)
// One block per (n, group); groups=32 -> 8 channels per group.
// ---------------------------------------------------------------------------
__global__ void groupnorm_tokens(const float* __restrict__ Y, const float* __restrict__ g,
                                 const float* __restrict__ bb, float* __restrict__ s32,
                                 __bf16* __restrict__ sbf, int Pl, long tokBase, long Ltot) {
  const int n = blockIdx.x >> 5;
  const int c0 = (blockIdx.x & 31) * 8;
  const int tid = threadIdx.x;
  const int total = Pl * 8;
  float s = 0.f, q = 0.f;
  for (int i = tid; i < total; i += 256) {
    const int p = i >> 3, c = c0 + (i & 7);
    const float v = Y[((long)n * Pl + p) * 256 + c];
    s += v; q += v * v;
  }
  __shared__ float rs[256], rq[256];
  rs[tid] = s; rq[tid] = q;
  __syncthreads();
  for (int st = 128; st > 0; st >>= 1) {
    if (tid < st) { rs[tid] += rs[tid + st]; rq[tid] += rq[tid + st]; }
    __syncthreads();
  }
  const float mean = rs[0] / (float)total;
  const float var  = rq[0] / (float)total - mean * mean;
  const float rstd = rsqrtf(var + 1e-5f);
  for (int i = tid; i < total; i += 256) {
    const int p = i >> 3, c = c0 + (i & 7);
    const float v = Y[((long)n * Pl + p) * 256 + c];
    const float o = (v - mean) * rstd * g[c] + bb[c];
    const long dst = ((long)n * Ltot + tokBase + p) * 256 + c;
    s32[dst] = o;
    sbf[dst] = (__bf16)o;
  }
}

// ---------------------------------------------------------------------------
// softmax over 20 (NL*NP) attention-weight logits, in place
// ---------------------------------------------------------------------------
__global__ void softmax20(float* __restrict__ aw, long total) {
  long i = (long)blockIdx.x * 256 + threadIdx.x;
  if (i >= total) return;
  const long m = i >> 3;
  const int  h = (int)(i & 7);
  float* p = aw + m * 160 + h * 20;
  float mx = p[0];
#pragma unroll
  for (int j = 1; j < 20; ++j) mx = fmaxf(mx, p[j]);
  float e[20], sum = 0.f;
#pragma unroll
  for (int j = 0; j < 20; ++j) { e[j] = __expf(p[j] - mx); sum += e[j]; }
  const float inv = 1.f / sum;
#pragma unroll
  for (int j = 0; j < 20; ++j) p[j] = e[j] * inv;
}

// ---------------------------------------------------------------------------
// MS-deformable sampling: one wave per (token, head); lane = channel (DH=32),
// so each corner gather is a single coalesced 128B line per wave.
// ---------------------------------------------------------------------------
__device__ __forceinline__ float sampleV(const float* vl, int ix, int iy, int W, int H) {
  const bool valid = (ix >= 0) & (ix < W) & (iy >= 0) & (iy < H);
  const int cx = min(max(ix, 0), W - 1);
  const int cy = min(max(iy, 0), H - 1);
  const float v = vl[(long)(cy * W + cx) * 256];
  return valid ? v : 0.f;
}

__global__ void msdeform_kernel(const float* __restrict__ value, const float* __restrict__ ref,
                                const float* __restrict__ off, const float* __restrict__ aw,
                                __bf16* __restrict__ out, int Ltot, long Mtok) {
  const long gw = ((long)blockIdx.x * blockDim.x + threadIdx.x) >> 5;
  const int lane = threadIdx.x & 31;
  if (gw >= Mtok * 8) return;
  const long m = gw >> 3;
  const int  h = (int)(gw & 7);
  const int  n = (int)(m / Ltot);

  const int hs[5]  = {100, 50, 25, 13, 7};
  const int wss[5] = {100, 50, 25, 13, 7};
  const int stt[5] = {0, 10000, 12500, 13125, 13294};

  float acc = 0.f;
  const long vbaseN = (long)n * Ltot;
#pragma unroll
  for (int l = 0; l < 5; ++l) {
    const int H = hs[l], W = wss[l], st = stt[l];
    const float rx = ref[m * 10 + l * 2 + 0];
    const float ry = ref[m * 10 + l * 2 + 1];
    const float* vl = value + (vbaseN + st) * 256 + h * 32 + lane;
#pragma unroll
    for (int p = 0; p < 4; ++p) {
      const long ob = m * 320 + h * 40 + l * 8 + p * 2;
      const float lx = rx + off[ob]     / (float)W;
      const float ly = ry + off[ob + 1] / (float)H;
      const float a  = aw[m * 160 + h * 20 + l * 4 + p];
      const float xf = lx * W - 0.5f, yf = ly * H - 0.5f;
      const float x0f = floorf(xf), y0f = floorf(yf);
      const float wx = xf - x0f, wy = yf - y0f;
      const int x0 = (int)x0f, y0 = (int)y0f;
      acc += sampleV(vl, x0,     y0,     W, H) * ((1.f - wx) * (1.f - wy) * a);
      acc += sampleV(vl, x0 + 1, y0,     W, H) * (wx * (1.f - wy) * a);
      acc += sampleV(vl, x0,     y0 + 1, W, H) * ((1.f - wx) * wy * a);
      acc += sampleV(vl, x0 + 1, y0 + 1, W, H) * (wx * wy * a);
    }
  }
  out[m * 256 + h * 32 + lane] = (__bf16)acc;
}

// ---------------------------------------------------------------------------
// residual add + LayerNorm over D=256: one wave per token, 8 channels/lane
// ---------------------------------------------------------------------------
__global__ void add_layernorm(const float* __restrict__ X, const float* __restrict__ Yv,
                              const float* __restrict__ g, const float* __restrict__ bb,
                              float* __restrict__ o32, __bf16* __restrict__ obf, long Mtok) {
  const long w = ((long)blockIdx.x * blockDim.x + threadIdx.x) >> 5;
  const int lane = threadIdx.x & 31;
  if (w >= Mtok) return;
  const float* xr = X  + w * 256;
  const float* yr = Yv + w * 256;
  float v[8], s = 0.f;
#pragma unroll
  for (int i = 0; i < 8; ++i) { const int c = lane * 8 + i; v[i] = xr[c] + yr[c]; s += v[i]; }
  for (int o = 16; o > 0; o >>= 1) s += __shfl_xor(s, o, 32);
  const float mean = s * (1.f / 256.f);
  float q = 0.f;
#pragma unroll
  for (int i = 0; i < 8; ++i) { const float d = v[i] - mean; q += d * d; }
  for (int o = 16; o > 0; o >>= 1) q += __shfl_xor(q, o, 32);
  const float rstd = rsqrtf(q * (1.f / 256.f) + 1e-5f);
#pragma unroll
  for (int i = 0; i < 8; ++i) {
    const int c = lane * 8 + i;
    const float o = (v[i] - mean) * rstd * g[c] + bb[c];
    o32[w * 256 + c] = o;
    obf[w * 256 + c] = (__bf16)o;
  }
}

// token-major h -> NCHW fp32 (identity level output)
__global__ void copy_tokens_to_nchw(const float* __restrict__ h, float* __restrict__ out,
                                    int Pl, long tokBase, long Ltot, long total) {
  long i = (long)blockIdx.x * 256 + threadIdx.x;
  if (i >= total) return;
  const int p = (int)(i % Pl);
  const int c = (int)((i / Pl) % 256);
  const int n = (int)(i / ((long)Pl * 256));
  out[i] = h[((long)n * Ltot + tokBase + p) * 256 + c];
}

// ---------------------------------------------------------------------------
// Host side
// ---------------------------------------------------------------------------
extern "C" void kernel_launch(void* const* d_in, const int* in_sizes, int n_in,
                              void* d_out, int out_size, void* d_ws, size_t ws_size,
                              hipStream_t stream) {
  (void)in_sizes; (void)n_in; (void)out_size; (void)ws_size;
  const float* x0   = (const float*)d_in[0];
  const float* x1   = (const float*)d_in[1];
  const float* x2   = (const float*)d_in[2];
  const float* x3   = (const float*)d_in[3];
  const float* W_ref = (const float*)d_in[4];  const float* b_ref = (const float*)d_in[5];
  const float* W_off = (const float*)d_in[6];  const float* b_off = (const float*)d_in[7];
  const float* W_aw  = (const float*)d_in[8];  const float* b_aw  = (const float*)d_in[9];
  const float* W_val = (const float*)d_in[10]; const float* b_val = (const float*)d_in[11];
  const float* W_out = (const float*)d_in[12]; const float* b_out = (const float*)d_in[13];
  const float* gn_g  = (const float*)d_in[14]; const float* gn_b  = (const float*)d_in[15];
  const float* Wc1   = (const float*)d_in[16]; const float* bc1   = (const float*)d_in[17];
  const float* Wc2   = (const float*)d_in[18]; const float* bc2   = (const float*)d_in[19];
  const float* Wc3   = (const float*)d_in[20]; const float* bc3   = (const float*)d_in[21];
  const float* Wc4   = (const float*)d_in[22]; const float* bc4   = (const float*)d_in[23];
  const float* Wc11  = (const float*)d_in[24]; const float* bc11  = (const float*)d_in[25];
  const float* Wc22  = (const float*)d_in[26]; const float* bc22  = (const float*)d_in[27];
  const float* Wc33  = (const float*)d_in[28]; const float* bc33  = (const float*)d_in[29];
  const float* ln1_g = (const float*)d_in[30]; const float* ln1_b = (const float*)d_in[31];
  const float* W1    = (const float*)d_in[32]; const float* b1f   = (const float*)d_in[33];
  const float* W2    = (const float*)d_in[34]; const float* b2f   = (const float*)d_in[35];
  const float* ln2_g = (const float*)d_in[36]; const float* ln2_b = (const float*)d_in[37];

  const int Nb = 2;
  const long L = 13343;                       // 10000+2500+625+169+49
  const long Mtok = Nb * L;                   // 26686
  const int  P[5]  = {10000, 2500, 625, 169, 49};
  const long S[5]  = {0, 10000, 12500, 13125, 13294};

  // ---- workspace carve-out ----
  char* wptr = (char*)d_ws;
  auto alloc = [&](size_t bytes) -> void* {
    void* p = (void*)wptr;
    wptr += (bytes + 255) & ~(size_t)255;
    return p;
  };
  float*  src32  = (float*) alloc(Mtok * 256 * 4);
  __bf16* srcbf  = (__bf16*)alloc(Mtok * 256 * 2);
  float*  value32= (float*) alloc(Mtok * 256 * 4);
  float*  off32  = (float*) alloc(Mtok * 320 * 4);
  float*  aw32   = (float*) alloc(Mtok * 160 * 4);
  float*  ref32  = (float*) alloc(Mtok * 10 * 4);
  __bf16* attnbf = (__bf16*)alloc(Mtok * 256 * 2);
  float*  attn32 = (float*) alloc(Mtok * 256 * 4);
  float*  h32    = (float*) alloc(Mtok * 256 * 4);
  __bf16* hbf    = (__bf16*)alloc(Mtok * 256 * 2);
  __bf16* ffn1bf = (__bf16*)alloc(Mtok * 1024 * 2);
  float*  ffn232 = (float*) alloc(Mtok * 256 * 4);
  float*  h232   = (float*) alloc(Mtok * 256 * 4);
  __bf16* h2bf   = (__bf16*)alloc(Mtok * 256 * 2);
  __bf16* X0bf   = (__bf16*)alloc((long)Nb * P[0] * 128 * 2);
  __bf16* X2bf   = (__bf16*)alloc((long)Nb * P[2] * 512 * 2);
  __bf16* X3bf   = (__bf16*)alloc((long)Nb * P[3] * 1024 * 2);
  float*  y0     = (float*) alloc((long)Nb * P[0] * 256 * 4);
  float*  y2     = (float*) alloc((long)Nb * P[2] * 256 * 4);
  float*  y3     = (float*) alloc((long)Nb * P[3] * 256 * 4);
  float*  y4     = (float*) alloc((long)Nb * P[4] * 256 * 4);
  __bf16* Wc1b   = (__bf16*)alloc(256 * 128 * 2);
  __bf16* Wc2b   = (__bf16*)alloc(256 * 512 * 2);
  __bf16* Wc3b   = (__bf16*)alloc(256 * 1024 * 2);
  __bf16* Wrefb  = (__bf16*)alloc(10 * 256 * 2);
  __bf16* Woffb  = (__bf16*)alloc(320 * 256 * 2);
  __bf16* Wawb   = (__bf16*)alloc(160 * 256 * 2);
  __bf16* Wvalb  = (__bf16*)alloc(256 * 256 * 2);
  __bf16* Woutb  = (__bf16*)alloc(256 * 256 * 2);
  __bf16* W1b    = (__bf16*)alloc(1024 * 256 * 2);
  __bf16* W2b    = (__bf16*)alloc(256 * 1024 * 2);
  __bf16* Wc11b  = (__bf16*)alloc(128 * 256 * 2);
  __bf16* Wc22b  = (__bf16*)alloc(512 * 256 * 2);
  __bf16* Wc33b  = (__bf16*)alloc(1024 * 256 * 2);

  auto cvt = [&](const float* s, __bf16* d, long n) {
    cvt_f32_bf16<<<(unsigned)((n + 255) / 256), 256, 0, stream>>>(s, d, n);
  };
  // Pl==0 means "identity row map": pass Pl=M, Ltot=0, tokBase=0 so the
  // kernel's single row-map formula degenerates to arow = m with no branch.
  auto gemm = [&](const __bf16* A, const __bf16* Bw, const float* bias,
                  float* C32, __bf16* Cbf, float* Cn,
                  int M, int Ncol, int K, int act, int Pl, long tokBase) {
    const int  rowPl = (Pl > 0) ? Pl : M;
    const long lt    = (Pl > 0) ? L : 0;
    const long tb    = (Pl > 0) ? tokBase : 0;
    dim3 g((M + BM - 1) / BM, (Ncol + BN - 1) / BN);
    gemm_wmma_bf16<<<g, 128, 0, stream>>>(A, Bw, bias, C32, Cbf, Cn,
                                          M, Ncol, K, act, rowPl, tb, lt);
  };

  // 1. weights -> bf16
  cvt(Wc1,  Wc1b,  256 * 128);   cvt(Wc2,  Wc2b,  256 * 512);
  cvt(Wc3,  Wc3b,  256 * 1024);  cvt(W_ref, Wrefb, 10 * 256);
  cvt(W_off, Woffb, 320 * 256);  cvt(W_aw, Wawb,  160 * 256);
  cvt(W_val, Wvalb, 256 * 256);  cvt(W_out, Woutb, 256 * 256);
  cvt(W1,   W1b,   1024 * 256);  cvt(W2,   W2b,   256 * 1024);
  cvt(Wc11, Wc11b, 128 * 256);   cvt(Wc22, Wc22b, 512 * 256);
  cvt(Wc33, Wc33b, 1024 * 256);

  // 2. NCHW -> token-major bf16 conv inputs
  {
    long t0 = (long)Nb * P[0] * 128;
    nchw_to_tok_bf16<<<(unsigned)((t0 + 255) / 256), 256, 0, stream>>>(x0, X0bf, 128, P[0], t0);
    long t2 = (long)Nb * P[2] * 512;
    nchw_to_tok_bf16<<<(unsigned)((t2 + 255) / 256), 256, 0, stream>>>(x2, X2bf, 512, P[2], t2);
    long t3 = (long)Nb * P[3] * 1024;
    nchw_to_tok_bf16<<<(unsigned)((t3 + 255) / 256), 256, 0, stream>>>(x3, X3bf, 1024, P[3], t3);
  }
  // 3. level-1 tokens straight into src
  {
    long t1 = (long)Nb * 2500 * 256;
    copy_x1_to_src<<<(unsigned)((t1 + 255) / 256), 256, 0, stream>>>(x1, src32, srcbf, S[1], L, t1);
  }
  // 4. lateral 1x1 convs as WMMA GEMMs + direct 3x3-s2
  gemm(X0bf, Wc1b, bc1, y0, nullptr, nullptr, Nb * P[0], 256, 128, 0, 0, 0);
  gemm(X2bf, Wc2b, bc2, y2, nullptr, nullptr, Nb * P[2], 256, 512, 0, 0, 0);
  gemm(X3bf, Wc3b, bc3, y3, nullptr, nullptr, Nb * P[3], 256, 1024, 0, 0, 0);
  conv3x3_s2<<<Nb * P[4], 256, 0, stream>>>(x3, Wc4, bc4, y4);

  // 5. GroupNorm each conv level into src
  groupnorm_tokens<<<Nb * 32, 256, 0, stream>>>(y0, gn_g, gn_b, src32, srcbf, P[0], S[0], L);
  groupnorm_tokens<<<Nb * 32, 256, 0, stream>>>(y2, gn_g, gn_b, src32, srcbf, P[2], S[2], L);
  groupnorm_tokens<<<Nb * 32, 256, 0, stream>>>(y3, gn_g, gn_b, src32, srcbf, P[3], S[3], L);
  groupnorm_tokens<<<Nb * 32, 256, 0, stream>>>(y4, gn_g, gn_b, src32, srcbf, P[4], S[4], L);

  // 6. projections (WMMA)
  gemm(srcbf, Wrefb, b_ref, ref32, nullptr, nullptr, (int)Mtok, 10, 256, 2, 0, 0);   // sigmoid
  gemm(srcbf, Wvalb, b_val, value32, nullptr, nullptr, (int)Mtok, 256, 256, 0, 0, 0);
  gemm(srcbf, Woffb, b_off, off32, nullptr, nullptr, (int)Mtok, 320, 256, 0, 0, 0);
  gemm(srcbf, Wawb,  b_aw,  aw32,  nullptr, nullptr, (int)Mtok, 160, 256, 0, 0, 0);
  {
    long t = Mtok * 8;
    softmax20<<<(unsigned)((t + 255) / 256), 256, 0, stream>>>(aw32, t);
  }

  // 7. deformable sampling (wave per (token,head), lane = channel)
  msdeform_kernel<<<(unsigned)Mtok, 256, 0, stream>>>(value32, ref32, off32, aw32, attnbf, (int)L, Mtok);

  // 8. output projection + LN1
  gemm(attnbf, Woutb, b_out, attn32, nullptr, nullptr, (int)Mtok, 256, 256, 0, 0, 0);
  add_layernorm<<<(unsigned)((Mtok * 32 + 255) / 256), 256, 0, stream>>>(
      src32, attn32, ln1_g, ln1_b, h32, hbf, Mtok);

  // 9. FFN (WMMA) + LN2
  gemm(hbf, W1b, b1f, nullptr, ffn1bf, nullptr, (int)Mtok, 1024, 256, 1, 0, 0);      // relu
  gemm(ffn1bf, W2b, b2f, ffn232, nullptr, nullptr, (int)Mtok, 256, 1024, 0, 0, 0);
  add_layernorm<<<(unsigned)((Mtok * 32 + 255) / 256), 256, 0, stream>>>(
      h32, ffn232, ln2_g, ln2_b, h232, h2bf, Mtok);

  // 10. output heads
  float* out  = (float*)d_out;
  float* out0 = out;                                     // (2,128,100,100)
  float* out1 = out0 + (long)Nb * 128 * P[0];            // (2,256,50,50)
  float* out2 = out1 + (long)Nb * 256 * P[1];            // (2,512,25,25)
  float* out3 = out2 + (long)Nb * 512 * P[2];            // (2,1024,13,13)

  gemm(h2bf, Wc11b, bc11, nullptr, nullptr, out0, Nb * P[0], 128, 256, 0, P[0], S[0]);
  {
    long t = (long)Nb * 256 * P[1];
    copy_tokens_to_nchw<<<(unsigned)((t + 255) / 256), 256, 0, stream>>>(h232, out1, P[1], S[1], L, t);
  }
  gemm(h2bf, Wc22b, bc22, nullptr, nullptr, out2, Nb * P[2], 512, 256, 0, P[2], S[2]);
  gemm(h2bf, Wc33b, bc33, nullptr, nullptr, out3, Nb * P[3], 1024, 256, 0, P[3], S[3]);
}
